// RiemannRotaryAttention_11733850653066
// MI455X (gfx1250) — matrix operand order repaired and checked by reading, not compiled, not run on hardware
//
#include <hip/hip_runtime.h>

// ---------------------------------------------------------------------------
// RiemannRotaryAttention for MI455X (gfx1250, wave32, WMMA bf16 path)
// b=4, n=2048, D_MODEL=1024, H=16, Dh=64, NPAIRS=32
// ---------------------------------------------------------------------------

#define B_SZ     4
#define N_SEQ    2048
#define DMODEL   1024
#define NHEADS   16
#define DHEAD    64
#define NPAIRS   32

typedef __attribute__((ext_vector_type(16))) __bf16 v16bf;
typedef __attribute__((ext_vector_type(8)))  float  v8f;

union FragBF {
  v16bf v;
  uint4 q[2];
  unsigned short u[16];
};

__constant__ float GAMMA[NPAIRS] = {
  14.134725142f, 21.022039639f, 25.01085758f,  30.424876126f,
  32.935061588f, 37.586178159f, 40.918719012f, 43.327073281f,
  48.005150881f, 49.773832478f, 52.970321478f, 56.446247697f,
  59.347044003f, 60.831778525f, 65.112544048f, 67.079810529f,
  69.546401711f, 72.067157674f, 75.704690699f, 77.144840069f,
  79.33737502f,  82.910380854f, 84.735492981f, 87.425274613f,
  88.809111208f, 92.491899271f, 94.651344041f, 95.870634228f,
  98.831194218f, 101.317851006f, 103.72553804f, 105.446623052f
};

__device__ __forceinline__ unsigned short f2bf_u16(float f) {
  union { float f; unsigned u; } v; v.f = f;
  unsigned r = v.u + 0x7FFFu + ((v.u >> 16) & 1u);
  return (unsigned short)(r >> 16);
}

// A-fragment (16x32, row-major source, ld = elements per row).
// Lane l in [0,16): row M=l, k 0..7 / 16..23; lane l+16: k 8..15 / 24..31.
__device__ __forceinline__ v16bf load_fragA_bf16(const unsigned short* __restrict__ base, int ld) {
  const int lane = threadIdx.x & 31;
  const int row  = lane & 15;
  const int kh   = lane >> 4;
  FragBF f;
  f.q[0] = *(const uint4*)(base + (size_t)row * ld + kh * 8);
  f.q[1] = *(const uint4*)(base + (size_t)row * ld + 16 + kh * 8);
  return f.v;
}

// B-fragment (32x16 K x N) read from a TRANSPOSED [N][K] tile (ld = K stride).
// Lane l in [0,16): col N=l, k 0..7 / 16..23; lane l+16: col N=l, k 8..15 / 24..31.
__device__ __forceinline__ v16bf load_fragB_bf16(const unsigned short* __restrict__ base, int ld) {
  const int lane = threadIdx.x & 31;
  const int col  = lane & 15;
  const int kh   = lane >> 4;
  FragBF f;
  f.q[0] = *(const uint4*)(base + (size_t)col * ld + kh * 8);
  f.q[1] = *(const uint4*)(base + (size_t)col * ld + 16 + kh * 8);
  return f.v;
}

#define WMMA_BF16(a, b, c) \
  __builtin_amdgcn_wmma_f32_16x16x32_bf16(false, (a), false, (b), (short)0, (c), false, false)

// ---------------------------------------------------------------------------
// fp32 -> bf16 convert
// ---------------------------------------------------------------------------
__global__ void k_f32_to_bf16(const float* __restrict__ in,
                              unsigned short* __restrict__ out, size_t n) {
  size_t i = (size_t)blockIdx.x * blockDim.x + threadIdx.x;
  if (i < n) out[i] = f2bf_u16(in[i]);
}

// ---------------------------------------------------------------------------
// bf16 GEMM: C(f32, MxN) = A(bf16, MxK row-major) * B(bf16, KxN row-major)
// 128x128 tile per 256-thread block (8 waves in 4x2, each wave 32x64).
// K-step 64; 16 WMMAs per wave per K-step with A x4 / B x2 fragment reuse.
// ---------------------------------------------------------------------------
__global__ __launch_bounds__(256) void k_gemm_bf16(
    const unsigned short* __restrict__ A, const unsigned short* __restrict__ Bm,
    float* __restrict__ C, int M, int N, int K) {
  __shared__ unsigned short As[128 * 72];   // [m][k], ld=72
  __shared__ unsigned short Bs[128 * 72];   // [n][k] (transposed), ld=72

  const int t    = threadIdx.x;
  const int wave = t >> 5;
  const int wr   = wave >> 1;   // 0..3 -> 32-row strip
  const int wc   = wave & 1;    // 0..1 -> 64-col strip
  const int m0   = blockIdx.y * 128;
  const int n0   = blockIdx.x * 128;

  v8f acc[2][4] = {};

  for (int k0 = 0; k0 < K; k0 += 64) {
    // Stage A tile 128x64 (4 x uint4 per thread)
#pragma unroll
    for (int it = 0; it < 4; ++it) {
      int u = t + it * 256;
      int r = u >> 3, seg = (u & 7) * 8;
      uint4 d = *(const uint4*)(A + (size_t)(m0 + r) * K + k0 + seg);
      *(uint4*)(As + r * 72 + seg) = d;
    }
    // Stage B tile 64x128, store transposed [n][k]
#pragma unroll
    for (int it = 0; it < 4; ++it) {
      int u = t + it * 256;
      int kr = u >> 4, nc = (u & 15) * 8;
      uint4 d = *(const uint4*)(Bm + (size_t)(k0 + kr) * N + n0 + nc);
      const unsigned short* du = (const unsigned short*)&d;
#pragma unroll
      for (int i = 0; i < 8; ++i) Bs[(nc + i) * 72 + kr] = du[i];
    }
    // Prefetch next K-tile into L2 while this tile's WMMAs run
    if (k0 + 64 < K) {
      __builtin_prefetch(A + (size_t)(m0 + (t & 127)) * K + (k0 + 64), 0, 1);
      __builtin_prefetch(Bm + (size_t)(k0 + 64 + (t & 63)) * N + n0 + ((t >> 6) & 1) * 64, 0, 1);
    }
    __syncthreads();

    v16bf a[2][2];
#pragma unroll
    for (int mi = 0; mi < 2; ++mi) {
      a[mi][0] = load_fragA_bf16(As + (wr * 32 + mi * 16) * 72, 72);
      a[mi][1] = load_fragA_bf16(As + (wr * 32 + mi * 16) * 72 + 32, 72);
    }
#pragma unroll
    for (int tn = 0; tn < 4; ++tn) {
      v16bf b0 = load_fragB_bf16(Bs + (wc * 64 + tn * 16) * 72, 72);
      v16bf b1 = load_fragB_bf16(Bs + (wc * 64 + tn * 16) * 72 + 32, 72);
      acc[0][tn] = WMMA_BF16(a[0][0], b0, acc[0][tn]);
      acc[0][tn] = WMMA_BF16(a[0][1], b1, acc[0][tn]);
      acc[1][tn] = WMMA_BF16(a[1][0], b0, acc[1][tn]);
      acc[1][tn] = WMMA_BF16(a[1][1], b1, acc[1][tn]);
    }
    __syncthreads();
  }

  const int lane = t & 31;
  const int kh   = lane >> 4;
  const int col  = lane & 15;
#pragma unroll
  for (int mi = 0; mi < 2; ++mi) {
#pragma unroll
    for (int tn = 0; tn < 4; ++tn) {
#pragma unroll
      for (int i = 0; i < 8; ++i) {
        int m = m0 + wr * 32 + mi * 16 + i + 8 * kh;
        C[(size_t)m * N + n0 + wc * 64 + tn * 16 + col] = acc[mi][tn][i];
      }
    }
  }
}

// ---------------------------------------------------------------------------
// Rotary + pack: qkv f32 (b,n,3,h,64) -> q,k rotated + v, bf16 in (b,h,n,64)
// One thread per (b,h,pos,pair).
// ---------------------------------------------------------------------------
__global__ void k_rotary_pack(const float* __restrict__ qkv,
                              const float* __restrict__ log_scale,
                              unsigned short* __restrict__ qb,
                              unsigned short* __restrict__ kb,
                              unsigned short* __restrict__ vb) {
  int idx = blockIdx.x * blockDim.x + threadIdx.x;   // 22 bits used
  int p   = idx & 31;
  int pos = (idx >> 5) & 2047;
  int h   = (idx >> 16) & 15;
  int b   = idx >> 20;

  float inv   = GAMMA[0] / GAMMA[p];
  float theta = (float)pos * inv * expf(log_scale[h]);
  float c = cosf(theta), s = sinf(theta);

  size_t rowbase = ((size_t)b * N_SEQ + pos) * (3 * DMODEL);
  int colq = h * DHEAD + 2 * p;
  float q1 = qkv[rowbase + colq],             q2 = qkv[rowbase + colq + 1];
  float k1 = qkv[rowbase + DMODEL + colq],    k2 = qkv[rowbase + DMODEL + colq + 1];
  float v1 = qkv[rowbase + 2*DMODEL + colq],  v2 = qkv[rowbase + 2*DMODEL + colq + 1];

  size_t obase = ((((size_t)b * NHEADS + h) * N_SEQ) + pos) * DHEAD + 2 * p;
  qb[obase]     = f2bf_u16(q1 * c - q2 * s);
  qb[obase + 1] = f2bf_u16(q1 * s + q2 * c);
  kb[obase]     = f2bf_u16(k1 * c - k2 * s);
  kb[obase + 1] = f2bf_u16(k1 * s + k2 * c);
  vb[obase]     = f2bf_u16(v1);
  vb[obase + 1] = f2bf_u16(v2);
}

// ---------------------------------------------------------------------------
// Causal flash attention. Grid (n/64, H, B), block 128 (4 waves).
// Wave w owns 16 q-rows; 64-key tiles; online softmax in fp32.
// 16 WMMAs per key-tile (8 QK^T with K-dim 64, 8 PV over two 32-key chunks).
// Output written bf16 in (b, n, DMODEL) layout ready for the W_o GEMM.
// ---------------------------------------------------------------------------
__global__ __launch_bounds__(128) void k_flash_attn(
    const unsigned short* __restrict__ qb, const unsigned short* __restrict__ kb,
    const unsigned short* __restrict__ vb, unsigned short* __restrict__ aout) {
  __shared__ unsigned short Ks[64 * 72];       // [key][d], ld=72
  __shared__ unsigned short Vt[64 * 72];       // [d][key] transposed, ld=72
  __shared__ unsigned short Ps[4][16 * 72];    // per-wave P scratch [row][key], ld=72

  const int t    = threadIdx.x;
  const int wave = t >> 5;
  const int lane = t & 31;
  const int kh   = lane >> 4;
  const int col  = lane & 15;
  const int qblk = blockIdx.x;
  const int h    = blockIdx.y;
  const int b    = blockIdx.z;
  const size_t bh = (((size_t)b * NHEADS + h) * N_SEQ) * DHEAD;
  const int Q0   = qblk * 64 + wave * 16;

  // Q fragments for this wave's 16 rows (k-chunks 0..31 / 32..63)
  const unsigned short* qbase = qb + bh + (size_t)Q0 * DHEAD;
  v16bf qa0 = load_fragA_bf16(qbase, DHEAD);
  v16bf qa1 = load_fragA_bf16(qbase + 32, DHEAD);

  float mrow[8], lrow[8];
  v8f o[4] = {};
#pragma unroll
  for (int i = 0; i < 8; ++i) { mrow[i] = -1e30f; lrow[i] = 0.f; }

  const int ntiles = qblk + 1;   // 64-key tiles covering keys 0 .. qblk*64+63
  for (int j = 0; j < ntiles; ++j) {
    const int j0 = j * 64;
    // Stage K tile (64x64) natural layout: 512 uint4, 4 per thread
#pragma unroll
    for (int it = 0; it < 4; ++it) {
      int u = t + it * 128;
      int kr = u >> 3, seg = (u & 7) * 8;
      uint4 d = *(const uint4*)(kb + bh + (size_t)(j0 + kr) * DHEAD + seg);
      *(uint4*)(Ks + kr * 72 + seg) = d;
    }
    // Stage V tile transposed -> Vt[d][key]
#pragma unroll
    for (int it = 0; it < 4; ++it) {
      int u = t + it * 128;
      int kr = u >> 3, seg = (u & 7) * 8;
      uint4 d = *(const uint4*)(vb + bh + (size_t)(j0 + kr) * DHEAD + seg);
      const unsigned short* du = (const unsigned short*)&d;
#pragma unroll
      for (int i = 0; i < 8; ++i) Vt[(seg + i) * 72 + kr] = du[i];
    }
    __syncthreads();

    // S = Q * K^T : four 16x16 tiles over 64 keys, K-dim = 64
    v8f s[4] = {};
#pragma unroll
    for (int tn = 0; tn < 4; ++tn) {
      v16bf b0 = load_fragB_bf16(Ks + tn * 16 * 72, 72);
      v16bf b1 = load_fragB_bf16(Ks + tn * 16 * 72 + 32, 72);
      s[tn] = WMMA_BF16(qa0, b0, s[tn]);
      s[tn] = WMMA_BF16(qa1, b1, s[tn]);
    }

    // Online softmax update (rows spread over lane-halves, cols over 16 lanes)
    float p[4][8];
#pragma unroll
    for (int i = 0; i < 8; ++i) {
      int qrow = Q0 + i + 8 * kh;
      float v[4];
#pragma unroll
      for (int tn = 0; tn < 4; ++tn) {
        v[tn] = s[tn][i] * 0.125f;
        if (j0 + tn * 16 + col > qrow) v[tn] = -1e30f;
      }
      float rmax = fmaxf(fmaxf(v[0], v[1]), fmaxf(v[2], v[3]));
#pragma unroll
      for (int off = 8; off >= 1; off >>= 1)
        rmax = fmaxf(rmax, __shfl_xor(rmax, off, 32));
      float mnew = fmaxf(mrow[i], rmax);
      float corr = expf(mrow[i] - mnew);
      float rsum = 0.f;
#pragma unroll
      for (int tn = 0; tn < 4; ++tn) {
        float e = expf(v[tn] - mnew);
        p[tn][i] = e;
        rsum += e;
      }
#pragma unroll
      for (int off = 8; off >= 1; off >>= 1)
        rsum += __shfl_xor(rsum, off, 32);
      lrow[i] = lrow[i] * corr + rsum;
      mrow[i] = mnew;
      o[0][i] *= corr; o[1][i] *= corr; o[2][i] *= corr; o[3][i] *= corr;
    }

    // C-layout -> A-layout via per-wave LDS round trip (16x64 P tile)
    unsigned short* pw = Ps[wave];
#pragma unroll
    for (int i = 0; i < 8; ++i) {
      int r = i + 8 * kh;
#pragma unroll
      for (int tn = 0; tn < 4; ++tn)
        pw[r * 72 + tn * 16 + col] = f2bf_u16(p[tn][i]);
    }
    asm volatile("s_wait_dscnt 0" ::: "memory");  // wave-private LDS RAW

    v16bf pa0 = load_fragA_bf16(pw, 72);        // keys 0..31
    v16bf pa1 = load_fragA_bf16(pw + 32, 72);   // keys 32..63
#pragma unroll
    for (int td = 0; td < 4; ++td) {
      v16bf bv0 = load_fragB_bf16(Vt + td * 16 * 72, 72);
      v16bf bv1 = load_fragB_bf16(Vt + td * 16 * 72 + 32, 72);
      o[td] = WMMA_BF16(pa0, bv0, o[td]);
      o[td] = WMMA_BF16(pa1, bv1, o[td]);
    }
    __syncthreads();
  }

  // Normalize and write bf16 output in (b, n, DMODEL) layout
#pragma unroll
  for (int i = 0; i < 8; ++i) {
    int qrow = Q0 + i + 8 * kh;
    float inv = 1.0f / lrow[i];
    size_t ro = ((size_t)b * N_SEQ + qrow) * DMODEL + h * DHEAD + col;
    aout[ro]      = f2bf_u16(o[0][i] * inv);
    aout[ro + 16] = f2bf_u16(o[1][i] * inv);
    aout[ro + 32] = f2bf_u16(o[2][i] * inv);
    aout[ro + 48] = f2bf_u16(o[3][i] * inv);
  }
}

// ---------------------------------------------------------------------------
// Launcher
// ---------------------------------------------------------------------------
extern "C" void kernel_launch(void* const* d_in, const int* in_sizes, int n_in,
                              void* d_out, int out_size, void* d_ws, size_t ws_size,
                              hipStream_t stream) {
  (void)in_sizes; (void)n_in; (void)out_size; (void)ws_size;

  const float* x         = (const float*)d_in[0];
  const float* w_qkv     = (const float*)d_in[1];
  const float* w_o       = (const float*)d_in[2];
  const float* log_scale = (const float*)d_in[3];
  float* out = (float*)d_out;

  const size_t BN   = (size_t)B_SZ * N_SEQ;            // 8192 rows
  const size_t nx   = BN * DMODEL;                     // 8,388,608
  const size_t nwq  = (size_t)DMODEL * 3 * DMODEL;     // 3,145,728
  const size_t nwo  = (size_t)DMODEL * DMODEL;         // 1,048,576
  const size_t nqkv = BN * 3 * DMODEL;                 // 25,165,824
  const size_t nhd  = BN * DMODEL;                     // per q/k/v buffer (b,h,n,64)

  char* ws = (char*)d_ws;
  unsigned short* xb    = (unsigned short*)ws;  ws += nx  * 2;
  unsigned short* wqkvb = (unsigned short*)ws;  ws += nwq * 2;
  unsigned short* wob   = (unsigned short*)ws;  ws += nwo * 2;
  float*          qkv   = (float*)ws;           ws += nqkv * 4;
  unsigned short* qb    = (unsigned short*)ws;  ws += nhd * 2;
  unsigned short* kbuf  = (unsigned short*)ws;  ws += nhd * 2;
  unsigned short* vbuf  = (unsigned short*)ws;  ws += nhd * 2;
  // qkv f32 is dead after rotary; reuse its storage for the attention output
  unsigned short* aout  = (unsigned short*)qkv;

  // 1) converts
  k_f32_to_bf16<<<dim3((unsigned)((nx  + 255) / 256)), 256, 0, stream>>>(x, xb, nx);
  k_f32_to_bf16<<<dim3((unsigned)((nwq + 255) / 256)), 256, 0, stream>>>(w_qkv, wqkvb, nwq);
  k_f32_to_bf16<<<dim3((unsigned)((nwo + 255) / 256)), 256, 0, stream>>>(w_o, wob, nwo);

  // 2) qkv = x @ w_qkv   (M=8192, N=3072, K=1024)
  k_gemm_bf16<<<dim3(3 * DMODEL / 128, (unsigned)(BN / 128)), 256, 0, stream>>>(
      xb, wqkvb, qkv, (int)BN, 3 * DMODEL, DMODEL);

  // 3) rotary + pack to (b,h,n,64) bf16
  {
    const size_t tot = (size_t)B_SZ * NHEADS * N_SEQ * NPAIRS;   // 4,194,304
    k_rotary_pack<<<dim3((unsigned)(tot / 256)), 256, 0, stream>>>(
        qkv, log_scale, qb, kbuf, vbuf);
  }

  // 4) causal flash attention
  k_flash_attn<<<dim3(N_SEQ / 64, NHEADS, B_SZ), 128, 0, stream>>>(qb, kbuf, vbuf, aout);

  // 5) out = attn @ w_o   (M=8192, N=1024, K=1024)
  k_gemm_bf16<<<dim3(DMODEL / 128, (unsigned)(BN / 128)), 256, 0, stream>>>(
      aout, wob, out, (int)BN, DMODEL, DMODEL);
}
